// TemporalGuidedModule_59356448031017
// MI455X (gfx1250) — matrix operand order repaired and checked by reading, not compiled, not run on hardware
//
#include <hip/hip_runtime.h>
#include <hip/hip_bf16.h>

typedef __attribute__((ext_vector_type(16))) _Float16 v16h;
typedef __attribute__((ext_vector_type(8)))  float    v8f;
typedef __attribute__((ext_vector_type(2)))  __fp16   fp16x2;

#define CDIM 256
#define NHEAD 8
#define NPNT 8
#define HDIM 32
#define HSZ 128
#define WSZ 128
#define HWQ (HSZ*WSZ)          // 16384
#define BATCH 2
#define MTOT (BATCH*HWQ)       // 32768

#define BM 128
#define BN 64
#define BK 32
#define SA_STRIDE 40           // halves; 80B row stride: 16B-aligned, bank-conflict-free
#define SA_STRIDE_U 20         // in 32-bit words

__device__ __forceinline__ unsigned pk2(float a, float b) {
  union { fp16x2 h; unsigned u; } c;
  c.h = __builtin_amdgcn_cvt_pkrtz(a, b);    // v_cvt_pk_rtz_f16_f32
  return c.u;
}

// ---------------------------------------------------------------------------
// f16-WMMA GEMM: Y[M,N] = A[M,K=256] * W[N,K]^T + bias
//   A_TRANS:     A element (m,k) read channels-first: A[b*C*HW + k*HW + p]
//   RELU:        apply max(0,.) in epilogue
//   STORE_TRANS: write Y as [B, C, H, W] instead of [M, N]
// ---------------------------------------------------------------------------
template<int N, bool A_TRANS, bool RELU, bool STORE_TRANS>
__global__ __launch_bounds__(256)
void tgm_gemm(const float* __restrict__ A, const float* __restrict__ W,
              const float* __restrict__ bias, float* __restrict__ Y)
{
  __shared__ __align__(16) unsigned sAu[BM * SA_STRIDE_U];
  __shared__ __align__(16) unsigned sBu[BN * SA_STRIDE_U];
  _Float16* sA = (_Float16*)sAu;

  const int tid    = threadIdx.x;
  const int wave   = tid >> 5;
  const int lane   = tid & 31;
  const int half16 = lane >> 4;
  const int r      = lane & 15;
  const int tileN  = blockIdx.x * BN;
  const int tileM  = blockIdx.y * BM;

  // batch/pixel decomposition (whole tile lies in one batch: 128 | HWQ)
  const int bIdx = tileM >> 14;            // / HWQ
  const int p0   = tileM & (HWQ - 1);

  v8f acc[4] = {};

  float4 areg[4];
  float4 breg[2];

  auto loadA = [&](int k0) {
    if (A_TRANS) {
      // vectorize along pixels: float4 covers m..m+3 for one k
      const float* Ab = A + (size_t)bIdx * CDIM * HWQ + p0;
      #pragma unroll
      for (int i = 0; i < 4; ++i) {
        int u  = tid + i * 256;            // 0..1023
        int k  = u >> 5;                   // 0..31
        int mg = u & 31;                   // float4 group along m
        areg[i] = *(const float4*)(Ab + (size_t)(k0 + k) * HWQ + 4 * mg);
      }
    } else {
      // vectorize along k: float4 covers k..k+3 for one row
      #pragma unroll
      for (int i = 0; i < 4; ++i) {
        int u   = tid + i * 256;           // 0..1023
        int row = u >> 3;                  // 0..127
        int cq  = u & 7;                   // float4 group along k
        areg[i] = *(const float4*)(A + (size_t)(tileM + row) * CDIM + k0 + 4 * cq);
      }
    }
  };
  auto loadB = [&](int k0) {
    #pragma unroll
    for (int i = 0; i < 2; ++i) {
      int u   = tid + i * 256;             // 0..511
      int row = u >> 3;                    // 0..63
      int cq  = u & 7;
      breg[i] = *(const float4*)(W + (size_t)(tileN + row) * CDIM + k0 + 4 * cq);
    }
  };
  auto storeTiles = [&]() {
    if (A_TRANS) {
      #pragma unroll
      for (int i = 0; i < 4; ++i) {
        int u  = tid + i * 256;
        int k  = u >> 5;
        int m0 = (u & 31) * 4;
        sA[(m0 + 0) * SA_STRIDE + k] = (_Float16)areg[i].x;
        sA[(m0 + 1) * SA_STRIDE + k] = (_Float16)areg[i].y;
        sA[(m0 + 2) * SA_STRIDE + k] = (_Float16)areg[i].z;
        sA[(m0 + 3) * SA_STRIDE + k] = (_Float16)areg[i].w;
      }
    } else {
      #pragma unroll
      for (int i = 0; i < 4; ++i) {
        int u   = tid + i * 256;
        int row = u >> 3;
        int cq  = u & 7;
        sAu[row * SA_STRIDE_U + 2 * cq + 0] = pk2(areg[i].x, areg[i].y);
        sAu[row * SA_STRIDE_U + 2 * cq + 1] = pk2(areg[i].z, areg[i].w);
      }
    }
    #pragma unroll
    for (int i = 0; i < 2; ++i) {
      int u   = tid + i * 256;
      int row = u >> 3;
      int cq  = u & 7;
      sBu[row * SA_STRIDE_U + 2 * cq + 0] = pk2(breg[i].x, breg[i].y);
      sBu[row * SA_STRIDE_U + 2 * cq + 1] = pk2(breg[i].z, breg[i].w);
    }
  };

  loadA(0);
  loadB(0);

  #pragma unroll
  for (int kt = 0; kt < CDIM / BK; ++kt) {
    storeTiles();
    __syncthreads();
    if (kt + 1 < CDIM / BK) {              // prefetch next K tile (overlaps WMMAs)
      loadA((kt + 1) * BK);
      loadB((kt + 1) * BK);
    }

    // A fragment: lane row = wave*16+r; half=0: K0..7 & K16..23 ; half=1: K8..15 & K24..31
    union { uint4 u[2]; v16h h; } af;
    const uint4* pa = (const uint4*)(&sA[(wave * 16 + r) * SA_STRIDE + half16 * 8]);
    af.u[0] = pa[0];
    af.u[1] = pa[2];

    #pragma unroll
    for (int j = 0; j < 4; ++j) {
      // B fragment: lane col = j*16+r; half=0: K0..7 & K16..23 ; half=1: K8..15 & K24..31
      union { uint4 u[2]; v16h h; } bf;
      const uint4* pb = (const uint4*)(((_Float16*)sBu) + (j * 16 + r) * SA_STRIDE + half16 * 8);
      bf.u[0] = pb[0];
      bf.u[1] = pb[2];
      acc[j] = __builtin_amdgcn_wmma_f32_16x16x32_f16(
          false, af.h, false, bf.h, (short)0, acc[j], false, false);
    }
    __syncthreads();
  }

  // epilogue: D layout VGPR v -> M = v + 8*half, lane -> N = r
  #pragma unroll
  for (int j = 0; j < 4; ++j) {
    int n = tileN + j * 16 + r;
    float bv = bias[n];
    #pragma unroll
    for (int v = 0; v < 8; ++v) {
      int m = tileM + wave * 16 + v + half16 * 8;
      float val = acc[j][v] + bv;
      if (RELU) val = fmaxf(val, 0.f);
      if (STORE_TRANS) {
        int b = m >> 14;
        int p = m & (HWQ - 1);
        Y[(size_t)b * CDIM * HWQ + (size_t)n * HWQ + p] = val;
      } else {
        Y[(size_t)m * N + n] = val;
      }
    }
  }
}

// ---------------------------------------------------------------------------
// MSDA core: one wave per (b, q, head); lane = head-dim channel.
// ---------------------------------------------------------------------------
__global__ __launch_bounds__(256)
void tgm_msda(const float* __restrict__ value, const float* __restrict__ offs,
              const float* __restrict__ alog, float* __restrict__ out)
{
  int gw   = blockIdx.x * 8 + (threadIdx.x >> 5);
  int lane = threadIdx.x & 31;           // d channel (HD=32)
  int h = gw & (NHEAD - 1);
  int q = (gw >> 3) & (HWQ - 1);
  int b = gw >> 17;

  int py = q >> 7;
  int px = q & (WSZ - 1);
  float refx = (px + 0.5f) * (1.f / WSZ);
  float refy = (py + 0.5f) * (1.f / HSZ);

  size_t base = ((size_t)(b * HWQ + q) * NHEAD + h);
  const float* ap = alog + base * NPNT;
  const float* op = offs + base * NPNT * 2;

  float wexp[NPNT];
  float mx = -1e30f;
  #pragma unroll
  for (int p = 0; p < NPNT; ++p) { wexp[p] = ap[p]; mx = fmaxf(mx, wexp[p]); }
  float s = 0.f;
  #pragma unroll
  for (int p = 0; p < NPNT; ++p) { wexp[p] = __expf(wexp[p] - mx); s += wexp[p]; }
  float inv = 1.f / s;

  const float* vb = value + (size_t)b * HWQ * CDIM + h * HDIM + lane;
  float acc = 0.f;

  #pragma unroll
  for (int p = 0; p < NPNT; ++p) {
    float aw = wexp[p] * inv;
    float ix = (refx + op[2 * p + 0] * (1.f / WSZ)) * WSZ - 0.5f;
    float iy = (refy + op[2 * p + 1] * (1.f / HSZ)) * HSZ - 0.5f;
    float x0f = floorf(ix), y0f = floorf(iy);
    int   x0 = (int)x0f,    y0 = (int)y0f;
    float fx = ix - x0f,    fy = iy - y0f;
    float w00 = (1.f - fx) * (1.f - fy) * aw;
    float w10 = fx * (1.f - fy) * aw;
    float w01 = (1.f - fx) * fy * aw;
    float w11 = fx * fy * aw;

    auto corner = [&](int xi, int yi, float w) {
      if (xi >= 0 && xi < WSZ && yi >= 0 && yi < HSZ) {
        acc += w * vb[(size_t)(yi * WSZ + xi) * CDIM];
      }
    };
    corner(x0,     y0,     w00);
    corner(x0 + 1, y0,     w10);
    corner(x0,     y0 + 1, w01);
    corner(x0 + 1, y0 + 1, w11);
  }
  out[(size_t)(b * HWQ + q) * CDIM + h * HDIM + lane] = acc;
}

// ---------------------------------------------------------------------------
__global__ __launch_bounds__(256)
void tgm_ew_mul(const float* __restrict__ a, const float* __restrict__ b,
                float* __restrict__ y, int n)
{
  int i = blockIdx.x * 256 + threadIdx.x;
  if (i < n) y[i] = a[i] * b[i];
}

__global__ __launch_bounds__(256)
void tgm_ew_add(const float* __restrict__ a, const float* __restrict__ b,
                float* __restrict__ y, int n)
{
  int i = blockIdx.x * 256 + threadIdx.x;
  if (i < n) y[i] = a[i] + b[i];
}

// ---------------------------------------------------------------------------
extern "C" void kernel_launch(void* const* d_in, const int* in_sizes, int n_in,
                              void* d_out, int out_size, void* d_ws, size_t ws_size,
                              hipStream_t stream) {
  const float* xt     = (const float*)d_in[0];
  const float* xt_1   = (const float*)d_in[1];
  const float* W_in1  = (const float*)d_in[2];
  const float* b_in1  = (const float*)d_in[3];
  const float* W_in2  = (const float*)d_in[4];
  const float* b_in2  = (const float*)d_in[5];
  const float* Wv     = (const float*)d_in[6];
  const float* bv     = (const float*)d_in[7];
  const float* Ws     = (const float*)d_in[8];
  const float* bs_off = (const float*)d_in[9];
  const float* Wa     = (const float*)d_in[10];
  const float* ba     = (const float*)d_in[11];
  const float* Wo     = (const float*)d_in[12];
  const float* bo     = (const float*)d_in[13];
  const float* Wt1    = (const float*)d_in[14];
  const float* bt1    = (const float*)d_in[15];
  const float* Wt2    = (const float*)d_in[16];
  const float* bt2    = (const float*)d_in[17];
  const float* Wout   = (const float*)d_in[18];
  const float* bout   = (const float*)d_in[19];

  const size_t MC = (size_t)MTOT * CDIM;          // 8,388,608 floats
  float* ws   = (float*)d_ws;
  float* x1   = ws;                               // [M,C] queries (live to the end)
  float* bufB = ws + MC;                          // x2 -> msout -> g -> tg
  float* bufC = ws + 2 * MC;                      // value -> xt_star -> t1 -> s
  float* offs = ws + 3 * MC;                      // [M, NH*NP*2]
  float* alog = ws + 3 * MC + (size_t)MTOT * NHEAD * NPNT * 2;  // [M, NH*NP]

  dim3 blk(256);
  const dim3 g256(CDIM / BN, MTOT / BM);          // (4, 256)
  const dim3 g128(128 / BN,  MTOT / BM);          // (2, 256)
  const dim3 g64 (64 / BN,   MTOT / BM);          // (1, 256)

  tgm_gemm<256, true,  false, false><<<g256, blk, 0, stream>>>(xt,   W_in1, b_in1, x1);
  tgm_gemm<256, true,  false, false><<<g256, blk, 0, stream>>>(xt_1, W_in2, b_in2, bufB);
  tgm_gemm<256, false, false, false><<<g256, blk, 0, stream>>>(bufB, Wv, bv, bufC);
  tgm_gemm<128, false, false, false><<<g128, blk, 0, stream>>>(x1, Ws, bs_off, offs);
  tgm_gemm<64,  false, false, false><<<g64,  blk, 0, stream>>>(x1, Wa, ba, alog);
  tgm_msda<<<MTOT * NHEAD / 8, blk, 0, stream>>>(bufC, offs, alog, bufB);
  tgm_gemm<256, false, false, false><<<g256, blk, 0, stream>>>(bufB, Wo, bo, bufC);
  tgm_ew_mul<<<(int)(MC / 256), blk, 0, stream>>>(x1, bufC, bufB, (int)MC);
  tgm_gemm<256, false, true,  false><<<g256, blk, 0, stream>>>(bufB, Wt1, bt1, bufC);
  tgm_gemm<256, false, false, false><<<g256, blk, 0, stream>>>(bufC, Wt2, bt2, bufB);
  tgm_ew_add<<<(int)(MC / 256), blk, 0, stream>>>(x1, bufB, bufC, (int)MC);
  tgm_gemm<256, false, true,  true ><<<g256, blk, 0, stream>>>(bufC, Wout, bout, (float*)d_out);
}
